// GraphLearning_3994319585588
// MI455X (gfx1250) — compile-verified, matrix operand order checked by reading, not run on hardware
//
#include <hip/hip_runtime.h>
#include <hip/hip_bf16.h>
#include <cstdint>

typedef __attribute__((ext_vector_type(16))) _Float16 v16h;
typedef __attribute__((ext_vector_type(8)))  _Float16 v8h;
typedef __attribute__((ext_vector_type(8)))  float    v8f;

#define B_SZ 512
#define L_SZ 64
#define H_SZ 1024

// ---------------------------------------------------------------------------
// WMMA helpers (CDNA5 wave32, V_WMMA_F32_16X16X32_F16)
// A/B frag layout (16-bit, 16x32): lane = row (A) / col (B) index (lane&15);
// lanes 0-15: K = k0+0..7 (halves 0..7), k0+16..23 (halves 8..15);
// lanes 16-31: +8 in K. Two contiguous 16B loads per lane from row-major
// (rows x K); B-frags load identically from B^T stored row-major.
// ---------------------------------------------------------------------------
__device__ __forceinline__ v16h load_frag(const _Float16* __restrict__ base,
                                          int ld, int row0, int k0) {
  int lane = threadIdx.x & 31;
  const _Float16* p = base + (size_t)(row0 + (lane & 15)) * ld
                           + (k0 + ((lane >> 4) << 3));
  v8h lo = *(const v8h*)(p);
  v8h hi = *(const v8h*)(p + 16);
  v16h f;
#pragma unroll
  for (int i = 0; i < 8; ++i) { f[i] = lo[i]; f[i + 8] = hi[i]; }
  return f;
}

__device__ __forceinline__ v8f wmma16(v16h a, v16h b, v8f c) {
  return __builtin_amdgcn_wmma_f32_16x16x32_f16(false, a, false, b,
                                                (short)0, c, false, false);
}

// D-matrix (16x16 f32, 8 VGPRs): lane l, vgpr v -> m = v + (l>>4)*8, n = l&15.

// ---------------------------------------------------------------------------
// Conversion / transpose kernels
// ---------------------------------------------------------------------------
__global__ void k_cvt(const float* __restrict__ in, _Float16* __restrict__ out,
                      size_t n) {
  size_t i = (size_t)blockIdx.x * blockDim.x + threadIdx.x;
  if (i < n) out[i] = (_Float16)in[i];
}

// x (B,L,H) f32 -> xh (L,B,H) f16  (per-timestep slices contiguous)
__global__ void k_cvt_x(const float* __restrict__ x, _Float16* __restrict__ xh) {
  size_t i = (size_t)blockIdx.x * blockDim.x + threadIdx.x;
  if (i >= (size_t)B_SZ * L_SZ * H_SZ) return;
  int h = (int)(i % H_SZ);
  size_t r = i / H_SZ;
  int b = (int)(r % B_SZ);
  int t = (int)(r / B_SZ);
  xh[i] = (_Float16)x[((size_t)b * L_SZ + t) * H_SZ + h];
}

// w (B,H,L) f32 -> wT (B,L,H) f16   (so B-frags load contiguously)
__global__ void k_cvt_w_t(const float* __restrict__ in, _Float16* __restrict__ out) {
  size_t i = (size_t)blockIdx.x * blockDim.x + threadIdx.x;
  if (i >= (size_t)B_SZ * L_SZ * H_SZ) return;
  int h = (int)(i % H_SZ);
  size_t r = i / H_SZ;
  int l = (int)(r % L_SZ);
  int b = (int)(r / L_SZ);
  out[i] = (_Float16)in[((size_t)b * H_SZ + h) * L_SZ + l];
}

// ---------------------------------------------------------------------------
// Fused GRU step: each wave computes a 32(batch) x 16(hidden-col) region for
// all 6 gate GEMMs (input + hidden paths for r/z/n). B-frags (weight columns)
// are reused across both 16-row A-tiles. A-fragments are double-buffered in
// source: WMMAs consume the previous iteration's loads while this iteration's
// loads are in flight, so waits need not drain loadcnt to zero.
// ---------------------------------------------------------------------------
__global__ void __launch_bounds__(128)
k_gru_step(const _Float16* __restrict__ xh_t,    // (B,H) f16, slice at time t
           const _Float16* __restrict__ h16_in,  // (B,H) f16
           const float*    __restrict__ h32_in,  // (B,H) f32
           const _Float16* __restrict__ wih,     // (3H,H) f16
           const _Float16* __restrict__ whh,     // (3H,H) f16
           const float* __restrict__ b_ih, const float* __restrict__ b_hh,
           _Float16* __restrict__ h16_out, float* __restrict__ h32_out,
           _Float16* __restrict__ xo16,          // (B,L,H) f16
           int t) {
  int wave = threadIdx.x >> 5;
  int tile = blockIdx.x * 4 + wave;               // 16 * 64 = 1024 wave tiles
  const int NT = H_SZ / 16;                       // 64 col tiles
  int m0 = (tile / NT) * 32;                      // 32 batch rows per wave
  int j0 = (tile % NT) * 16;                      // 16 hidden cols
  int lane = threadIdx.x & 31;

  v8f air0 = {}, aiz0 = {}, ain0 = {}, ahr0 = {}, ahz0 = {}, ahn0 = {};
  v8f air1 = {}, aiz1 = {}, ain1 = {}, ahr1 = {}, ahz1 = {}, ahn1 = {};

  // prologue: first A-fragments
  v16h ax0 = load_frag(xh_t,   H_SZ, m0,      0);
  v16h ah0 = load_frag(h16_in, H_SZ, m0,      0);
  v16h ax1 = load_frag(xh_t,   H_SZ, m0 + 16, 0);
  v16h ah1 = load_frag(h16_in, H_SZ, m0 + 16, 0);

  for (int kk = 0; kk < H_SZ; kk += 32) {
    // weight B-frags for this K-chunk (shared by both A-tiles)
    v16h bri = load_frag(wih, H_SZ,             j0, kk);
    v16h bzi = load_frag(wih, H_SZ,     H_SZ + j0, kk);
    v16h bni = load_frag(wih, H_SZ, 2 * H_SZ + j0, kk);
    v16h brh = load_frag(whh, H_SZ,             j0, kk);
    v16h bzh = load_frag(whh, H_SZ,     H_SZ + j0, kk);
    v16h bnh = load_frag(whh, H_SZ, 2 * H_SZ + j0, kk);

    // branch-free WGP-scope prefetch of the streamed-once x operand
    __builtin_prefetch(xh_t + (size_t)(m0 + (lane & 15)) * H_SZ
                            + ((kk + 128) & (H_SZ - 1)), 0, 3);

    // issue next iteration's A-frag loads (wrapped in-bounds on last iter)
    int kn = (kk + 32) & (H_SZ - 1);
    v16h nax0 = load_frag(xh_t,   H_SZ, m0,      kn);
    v16h nah0 = load_frag(h16_in, H_SZ, m0,      kn);
    v16h nax1 = load_frag(xh_t,   H_SZ, m0 + 16, kn);
    v16h nah1 = load_frag(h16_in, H_SZ, m0 + 16, kn);

    // consume previous iteration's A-frags
    air0 = wmma16(ax0, bri, air0);
    aiz0 = wmma16(ax0, bzi, aiz0);
    ain0 = wmma16(ax0, bni, ain0);
    ahr0 = wmma16(ah0, brh, ahr0);
    ahz0 = wmma16(ah0, bzh, ahz0);
    ahn0 = wmma16(ah0, bnh, ahn0);
    air1 = wmma16(ax1, bri, air1);
    aiz1 = wmma16(ax1, bzi, aiz1);
    ain1 = wmma16(ax1, bni, ain1);
    ahr1 = wmma16(ah1, brh, ahr1);
    ahz1 = wmma16(ah1, bzh, ahz1);
    ahn1 = wmma16(ah1, bnh, ahn1);

    ax0 = nax0; ah0 = nah0; ax1 = nax1; ah1 = nah1;
  }

  int n = j0 + (lane & 15);
  float bir = b_ih[n], biz = b_ih[H_SZ + n], bin_ = b_ih[2 * H_SZ + n];
  float bhr = b_hh[n], bhz = b_hh[H_SZ + n], bhn = b_hh[2 * H_SZ + n];
#pragma unroll
  for (int half = 0; half < 2; ++half) {
    v8f air = half ? air1 : air0;
    v8f aiz = half ? aiz1 : aiz0;
    v8f ain = half ? ain1 : ain0;
    v8f ahr = half ? ahr1 : ahr0;
    v8f ahz = half ? ahz1 : ahz0;
    v8f ahn = half ? ahn1 : ahn0;
    int mbase = m0 + half * 16 + ((lane >> 4) << 3);
#pragma unroll
    for (int v = 0; v < 8; ++v) {
      int bb = mbase + v;
      float hp = h32_in[(size_t)bb * H_SZ + n];
      float r  = 1.f / (1.f + __expf(-(air[v] + bir + ahr[v] + bhr)));
      float z  = 1.f / (1.f + __expf(-(aiz[v] + biz + ahz[v] + bhz)));
      float nn = tanhf(ain[v] + bin_ + r * (ahn[v] + bhn));
      float hn = (1.f - z) * nn + z * hp;
      h32_out[(size_t)bb * H_SZ + n] = hn;
      _Float16 hh = (_Float16)hn;
      h16_out[(size_t)bb * H_SZ + n] = hh;
      xo16[((size_t)bb * L_SZ + t) * H_SZ + n] = hh;
    }
  }
}

// ---------------------------------------------------------------------------
// k/q/v projections: per (batch, which) the 64x64 result = xo16[b] @ wT[b]^T.
// 2x2 register blocking: each wave computes 32x32 (4 accs), 4 loads / 4 WMMAs.
// v is stored transposed (b, l, s) for the adj GEMM's B-frags.
// ---------------------------------------------------------------------------
__global__ void __launch_bounds__(128)
k_kqv(const _Float16* __restrict__ xo16,
      const _Float16* __restrict__ kwT, const _Float16* __restrict__ qwT,
      const _Float16* __restrict__ vwT,
      _Float16* __restrict__ k16, _Float16* __restrict__ q16,
      _Float16* __restrict__ vT16) {
  int wave = threadIdx.x >> 5;
  int gtile = blockIdx.x * 4 + wave;   // 512*3*4 wave tiles
  int tile = gtile & 3;
  int rest = gtile >> 2;
  int which = rest % 3;
  int b = rest / 3;
  int m0 = (tile >> 1) * 32, n0 = (tile & 1) * 32;
  const _Float16* A  = xo16 + (size_t)b * L_SZ * H_SZ;
  const _Float16* Wt = (which == 0 ? kwT : which == 1 ? qwT : vwT)
                       + (size_t)b * L_SZ * H_SZ;
  v8f a00 = {}, a01 = {}, a10 = {}, a11 = {};
  for (int kk = 0; kk < H_SZ; kk += 32) {
    v16h A0 = load_frag(A,  H_SZ, m0,      kk);
    v16h A1 = load_frag(A,  H_SZ, m0 + 16, kk);
    v16h B0 = load_frag(Wt, H_SZ, n0,      kk);
    v16h B1 = load_frag(Wt, H_SZ, n0 + 16, kk);
    a00 = wmma16(A0, B0, a00);
    a01 = wmma16(A0, B1, a01);
    a10 = wmma16(A1, B0, a10);
    a11 = wmma16(A1, B1, a11);
  }
  int lane = threadIdx.x & 31;
#pragma unroll
  for (int mi = 0; mi < 2; ++mi) {
#pragma unroll
    for (int ni = 0; ni < 2; ++ni) {
      v8f acc = mi ? (ni ? a11 : a10) : (ni ? a01 : a00);
      int n = n0 + ni * 16 + (lane & 15);
      int mb = m0 + mi * 16 + ((lane >> 4) << 3);
#pragma unroll
      for (int v = 0; v < 8; ++v) {
        int m = mb + v;
        _Float16 val = (_Float16)acc[v];
        if (which == 0)      k16[((size_t)b * L_SZ + m) * L_SZ + n] = val;
        else if (which == 1) q16[((size_t)b * L_SZ + m) * L_SZ + n] = val;
        else                 vT16[((size_t)b * L_SZ + n) * L_SZ + m] = val;
      }
    }
  }
}

// score[b,t,s] = sum_l q[b,t,l] * k[b,s,l]
__global__ void __launch_bounds__(128)
k_score(const _Float16* __restrict__ q16, const _Float16* __restrict__ k16,
        float* __restrict__ score) {
  int wave = threadIdx.x >> 5;
  int gtile = blockIdx.x * 4 + wave;   // 512*16 tiles
  int tile = gtile & 15;
  int b = gtile >> 4;
  int m0 = (tile >> 2) * 16, n0 = (tile & 3) * 16;
  const _Float16* A  = q16 + (size_t)b * L_SZ * L_SZ;
  const _Float16* Bt = k16 + (size_t)b * L_SZ * L_SZ;
  v8f acc = {};
  for (int kk = 0; kk < L_SZ; kk += 32)
    acc = wmma16(load_frag(A, L_SZ, m0, kk), load_frag(Bt, L_SZ, n0, kk), acc);
  int lane = threadIdx.x & 31;
  int n = n0 + (lane & 15);
  int mb = m0 + ((lane >> 4) << 3);
#pragma unroll
  for (int v = 0; v < 8; ++v)
    score[((size_t)b * L_SZ + mb + v) * L_SZ + n] = acc[v];
}

// softmax over dim t (axis 1) for each (b, s) column; emit f16 P
__global__ void k_softmax(const float* __restrict__ score,
                          _Float16* __restrict__ P16) {
  int idx = blockIdx.x * blockDim.x + threadIdx.x;  // b*64 + s
  if (idx >= B_SZ * L_SZ) return;
  int b = idx / L_SZ, s = idx % L_SZ;
  const float* col = score + (size_t)b * L_SZ * L_SZ + s;
  float mx = -3.0e38f;
  for (int tt = 0; tt < L_SZ; ++tt) mx = fmaxf(mx, col[(size_t)tt * L_SZ]);
  float sum = 0.f;
  for (int tt = 0; tt < L_SZ; ++tt) sum += __expf(col[(size_t)tt * L_SZ] - mx);
  float inv = 1.f / sum;
  _Float16* out = P16 + (size_t)b * L_SZ * L_SZ + s;
  for (int tt = 0; tt < L_SZ; ++tt)
    out[(size_t)tt * L_SZ] = (_Float16)(__expf(col[(size_t)tt * L_SZ] - mx) * inv);
}

// adj[b,t,l] = relu( sum_s P[b,t,s] * vT[b,l,s] )
__global__ void __launch_bounds__(128)
k_adj(const _Float16* __restrict__ P16, const _Float16* __restrict__ vT16,
      float* __restrict__ adj) {
  int wave = threadIdx.x >> 5;
  int gtile = blockIdx.x * 4 + wave;
  int tile = gtile & 15;
  int b = gtile >> 4;
  int m0 = (tile >> 2) * 16, n0 = (tile & 3) * 16;
  const _Float16* A  = P16  + (size_t)b * L_SZ * L_SZ;
  const _Float16* Bt = vT16 + (size_t)b * L_SZ * L_SZ;
  v8f acc = {};
  for (int kk = 0; kk < L_SZ; kk += 32)
    acc = wmma16(load_frag(A, L_SZ, m0, kk), load_frag(Bt, L_SZ, n0, kk), acc);
  int lane = threadIdx.x & 31;
  int n = n0 + (lane & 15);
  int mb = m0 + ((lane >> 4) << 3);
#pragma unroll
  for (int v = 0; v < 8; ++v)
    adj[((size_t)b * L_SZ + mb + v) * L_SZ + n] = fmaxf(acc[v], 0.f);
}

// symmetrize + normalized Laplacian per batch (64x64 fits in LDS)
__global__ void __launch_bounds__(256)
k_final(const float* __restrict__ adj, float* __restrict__ out) {
  __shared__ float sA[L_SZ * L_SZ];
  __shared__ float sD[L_SZ];
  int b = blockIdx.x;
  const float* a = adj + (size_t)b * L_SZ * L_SZ;
  for (int i = threadIdx.x; i < L_SZ * L_SZ; i += blockDim.x) {
    int r = i >> 6, c = i & 63;
    sA[i] = 0.5f * (a[i] + a[c * L_SZ + r]);
  }
  __syncthreads();
  if (threadIdx.x < L_SZ) {
    float s = 0.f;
    for (int r = 0; r < L_SZ; ++r) s += sA[r * L_SZ + threadIdx.x];
    sD[threadIdx.x] = (s > 0.f) ? rsqrtf(s) : 0.f;
  }
  __syncthreads();
  float* o = out + (size_t)b * L_SZ * L_SZ;
  for (int i = threadIdx.x; i < L_SZ * L_SZ; i += blockDim.x) {
    int r = i >> 6, c = i & 63;
    o[i] = sD[r] * sA[i] * sD[c];
  }
}

// ---------------------------------------------------------------------------
// Host launcher
// ---------------------------------------------------------------------------
extern "C" void kernel_launch(void* const* d_in, const int* in_sizes, int n_in,
                              void* d_out, int out_size, void* d_ws, size_t ws_size,
                              hipStream_t stream) {
  const float* x    = (const float*)d_in[0];
  const float* w_ih = (const float*)d_in[1];
  const float* w_hh = (const float*)d_in[2];
  const float* b_ih = (const float*)d_in[3];
  const float* b_hh = (const float*)d_in[4];
  const float* kw   = (const float*)d_in[5];
  const float* qw   = (const float*)d_in[6];
  const float* vw   = (const float*)d_in[7];
  float* out = (float*)d_out;

  // workspace carve-up (bytes)
  char* ws = (char*)d_ws;
  size_t off = 0;
  auto take = [&](size_t bytes) { char* p = ws + off; off = (off + bytes + 255) & ~(size_t)255; return p; };
  const size_t W3H = (size_t)3 * H_SZ * H_SZ;
  const size_t BLH = (size_t)B_SZ * L_SZ * H_SZ;
  const size_t BH  = (size_t)B_SZ * H_SZ;
  const size_t BLL = (size_t)B_SZ * L_SZ * L_SZ;

  _Float16* wih16 = (_Float16*)take(W3H * 2);
  _Float16* whh16 = (_Float16*)take(W3H * 2);
  _Float16* xh    = (_Float16*)take(BLH * 2);   // (L,B,H)
  _Float16* h16a  = (_Float16*)take(BH * 2);
  _Float16* h16b  = (_Float16*)take(BH * 2);
  float*    h32a  = (float*)   take(BH * 4);
  float*    h32b  = (float*)   take(BH * 4);
  _Float16* xo16  = (_Float16*)take(BLH * 2);   // (B,L,H)
  _Float16* kwT   = (_Float16*)take(BLH * 2);   // (B,L,H)
  _Float16* qwT   = (_Float16*)take(BLH * 2);
  _Float16* vwT   = (_Float16*)take(BLH * 2);
  _Float16* k16   = (_Float16*)take(BLL * 2);
  _Float16* q16   = (_Float16*)take(BLL * 2);
  _Float16* vT16  = (_Float16*)take(BLL * 2);   // (b,l,s)
  _Float16* P16   = (_Float16*)take(BLL * 2);
  float*    sc32  = (float*)   take(BLL * 4);
  float*    adj32 = (float*)   take(BLL * 4);
  (void)ws_size; (void)in_sizes; (void)n_in; (void)out_size;

  // 1) precision conversions / transposes
  k_cvt<<<(unsigned)((W3H + 255) / 256), 256, 0, stream>>>(w_ih, wih16, W3H);
  k_cvt<<<(unsigned)((W3H + 255) / 256), 256, 0, stream>>>(w_hh, whh16, W3H);
  k_cvt_x<<<(unsigned)((BLH + 255) / 256), 256, 0, stream>>>(x, xh);
  k_cvt_w_t<<<(unsigned)((BLH + 255) / 256), 256, 0, stream>>>(kw, kwT);
  k_cvt_w_t<<<(unsigned)((BLH + 255) / 256), 256, 0, stream>>>(qw, qwT);
  k_cvt_w_t<<<(unsigned)((BLH + 255) / 256), 256, 0, stream>>>(vw, vwT);

  // 2) h0 = 0
  hipMemsetAsync(h16a, 0, BH * 2, stream);
  hipMemsetAsync(h32a, 0, BH * 4, stream);

  // 3) GRU scan: 64 sequential fused-step launches, ping-pong state buffers
  for (int t = 0; t < L_SZ; ++t) {
    const _Float16* hin16 = (t & 1) ? h16b : h16a;
    const float*    hin32 = (t & 1) ? h32b : h32a;
    _Float16* hout16      = (t & 1) ? h16a : h16b;
    float*    hout32      = (t & 1) ? h32a : h32b;
    k_gru_step<<<256, 128, 0, stream>>>(xh + (size_t)t * B_SZ * H_SZ,
                                        hin16, hin32, wih16, whh16,
                                        b_ih, b_hh, hout16, hout32, xo16, t);
  }

  // 4) attention pipeline
  k_kqv<<<(B_SZ * 3 * 4) / 4, 128, 0, stream>>>(xo16, kwT, qwT, vwT, k16, q16, vT16);
  k_score<<<(B_SZ * 16) / 4, 128, 0, stream>>>(q16, k16, sc32);
  k_softmax<<<(B_SZ * L_SZ + 255) / 256, 256, 0, stream>>>(sc32, P16);
  k_adj<<<(B_SZ * 16) / 4, 128, 0, stream>>>(P16, vT16, adj32);
  k_final<<<B_SZ, 256, 0, stream>>>(adj32, out);
}
